// NystromAttention2D_78134045048853
// MI455X (gfx1250) — matrix operand order repaired and checked by reading, not compiled, not run on hardware
//
#include <hip/hip_runtime.h>
#include <cstdint>

// ---------------------------------------------------------------------------
// Types for CDNA5 WMMA (wave32): V_WMMA_F32_16X16X32_BF16
// ---------------------------------------------------------------------------
typedef __attribute__((ext_vector_type(16))) __bf16 v16bf;
typedef __attribute__((ext_vector_type(8)))  __bf16 v8bf;
typedef __attribute__((ext_vector_type(8)))  float  v8f;

static __device__ __forceinline__ v16bf cat8(v8bf lo, v8bf hi) {
  return __builtin_shufflevector(lo, hi, 0,1,2,3,4,5,6,7,8,9,10,11,12,13,14,15);
}
static __device__ __forceinline__ v8f wmma_bf16(v16bf a, v16bf b, v8f c) {
  // D = A(16x32) * B(32x16) + C ; f32 accumulate
  return __builtin_amdgcn_wmma_f32_16x16x32_bf16(false, a, false, b, (short)0, c, false, false);
}

// Problem constants
#define NB   2
#define NN   16384
#define NDIM 512
#define NH   8
#define ND   64
#define NL   65
#define NLP  96     // L padded to multiple of 32 (for WMMA K) with zeros
#define NLP80 80    // L padded to multiple of 16 (sim3 / T3 row pad)

// ---------------------------------------------------------------------------
// Small helper kernels
// ---------------------------------------------------------------------------
__global__ void zero_f32_kernel(float* __restrict__ p, int n) {
  int t = blockIdx.x * 256 + threadIdx.x;
  if (t < n) p[t] = 0.f;
}

__global__ void cvt_f2bf_kernel(const float* __restrict__ src, __bf16* __restrict__ dst, int n) {
  int t = blockIdx.x * 256 + threadIdx.x;
  if (t < n) dst[t] = (__bf16)src[t];
}

// xp[b,i,:] = bf16(x[b, index[i], :])
__global__ void gather_cvt_kernel(const float* __restrict__ x, const int* __restrict__ index,
                                  __bf16* __restrict__ xp) {
  size_t t = (size_t)blockIdx.x * 256 + threadIdx.x;   // total NB*NN*NDIM = 16M
  int d = (int)(t & (NDIM - 1));
  int g = (int)(t >> 9);
  int b = g >> 14;
  int i = g & (NN - 1);
  xp[t] = (__bf16)x[((size_t)(b << 14) + index[i]) * NDIM + d];
}

// ---------------------------------------------------------------------------
// Generic bf16 WMMA GEMM:  C[M,N] = A[M,K] * B[K,N]
// Block: 256 threads (8 waves), tile 128x64, K-step 32.
// Wave (wm in 0..3, wn in 0..1) computes a 32x32 tile = 2x2 WMMA fragments.
// LDS: A tile row-major [128][32] (stride 40), B tile transposed [64][32]
// (stride 40) so each lane's fragment is contiguous 16B loads that match the
// CDNA5 bf16 A(16x32)/B(32x16) register layouts.
// mode 0: store bf16  |  mode 1: store f32  |  mode 2: f32 + bias + row scatter
// Batched over gridDim.z with two-level strides (z = z0*8 + z1).
// ---------------------------------------------------------------------------
__global__ __launch_bounds__(256) void gemm_bf16_kernel(
    const __bf16* __restrict__ A, const __bf16* __restrict__ B, void* __restrict__ Cv,
    int K, int lda, int ldb, int ldc,
    long sA0, long sA1, long sB0, long sB1, long sC0, long sC1,
    int mode, const float* __restrict__ bias, const int* __restrict__ scatter)
{
  __shared__ __bf16 At[128 * 40];
  __shared__ __bf16 Bt[64 * 40];
  const int tid = threadIdx.x, lane = tid & 31, wid = tid >> 5;
  const int wm = wid >> 1, wn = wid & 1;
  const int half = lane >> 4, lr = lane & 15;
  const int z = blockIdx.z, z0 = z >> 3, z1 = z & 7;
  A += (size_t)z0 * sA0 + (size_t)z1 * sA1;
  B += (size_t)z0 * sB0 + (size_t)z1 * sB1;
  const size_t coff = (size_t)z0 * sC0 + (size_t)z1 * sC1;
  const int row0 = blockIdx.y * 128, col0 = blockIdx.x * 64;

  v8f acc[2][2] = {};
  for (int k0 = 0; k0 < K; k0 += 32) {
    // stage A tile: 128 rows x 32 cols, 512 16-byte chunks
#pragma unroll
    for (int i = 0; i < 2; ++i) {
      int c = tid + i * 256;
      int r = c >> 2, cc = (c & 3) * 8;
      *(uint4*)&At[r * 40 + cc] = *(const uint4*)&A[(size_t)(row0 + r) * lda + k0 + cc];
    }
    // prefetch next A tile (emits global_prefetch_b8)
    if (k0 + 32 < K)
      __builtin_prefetch(&A[(size_t)(row0 + (tid >> 1)) * lda + k0 + 32], 0, 0);
    // stage B tile transposed: B[k0..k0+31][col0..col0+63] -> Bt[n][k]
    {
      int r = tid >> 3, cc = (tid & 7) * 8;
      v8bf v = *(const v8bf*)&B[(size_t)(k0 + r) * ldb + col0 + cc];
#pragma unroll
      for (int e = 0; e < 8; ++e) Bt[(cc + e) * 40 + r] = v[e];
    }
    __syncthreads();

    v16bf af[2], bfr[2];
#pragma unroll
    for (int im = 0; im < 2; ++im) {
      int r = wm * 32 + im * 16 + lr;
      af[im] = cat8(*(const v8bf*)&At[r * 40 + half * 8],
                    *(const v8bf*)&At[r * 40 + 16 + half * 8]);
    }
#pragma unroll
    for (int in = 0; in < 2; ++in) {
      int n = wn * 32 + in * 16 + lr;
      bfr[in] = cat8(*(const v8bf*)&Bt[n * 40 + half * 8],
                     *(const v8bf*)&Bt[n * 40 + 16 + half * 8]);
    }
#pragma unroll
    for (int im = 0; im < 2; ++im)
#pragma unroll
      for (int in = 0; in < 2; ++in)
        acc[im][in] = wmma_bf16(af[im], bfr[in], acc[im][in]);
    __syncthreads();
  }

  // epilogue: lane holds col (lr), rows j + 8*half
#pragma unroll
  for (int im = 0; im < 2; ++im)
#pragma unroll
    for (int in = 0; in < 2; ++in)
#pragma unroll
      for (int j = 0; j < 8; ++j) {
        int r = row0 + wm * 32 + im * 16 + half * 8 + j;
        int c = col0 + wn * 32 + in * 16 + lr;
        float v = acc[im][in][j];
        if (mode == 0) {
          ((__bf16*)Cv)[coff + (size_t)r * ldc + c] = (__bf16)v;
        } else if (mode == 1) {
          ((float*)Cv)[coff + (size_t)r * ldc + c] = v;
        } else {
          float vb = v + bias[c];
          int bb = r >> 14, jj = r & (NN - 1);
          int orow = (bb << 14) + scatter[jj];
          ((float*)Cv)[(size_t)orow * ldc + c] = vb;
        }
      }
}

// ---------------------------------------------------------------------------
// Landmark segment sums (segment_ids sorted): atomic flush on segment change.
// grid (16 bh, 64 chunks of 256 rows), block 256: thread = (d, row-group).
// ---------------------------------------------------------------------------
__global__ __launch_bounds__(256) void landmarks_kernel(
    const __bf16* __restrict__ qkv, const int* __restrict__ segids,
    float* __restrict__ qlf, float* __restrict__ klf)
{
  int bh = blockIdx.x, b = bh >> 3, h = bh & 7;
  int d = threadIdx.x & 63, rg = threadIdx.x >> 6;
  int base = blockIdx.y * 256;
  int cur = -1;
  float aq = 0.f, ak = 0.f;
  for (int r = base + rg; r < base + 256; r += 4) {
    int sg = segids[r];
    size_t off = (size_t)(b * NN + r) * (3 * NDIM) + h * ND + d;
    float qv = (float)qkv[off];
    float kv = (float)qkv[off + NDIM];
    if (sg != cur) {
      if (cur >= 0) {
        atomicAdd(&qlf[((size_t)bh * NL + cur) * ND + d], aq);
        atomicAdd(&klf[((size_t)bh * NL + cur) * ND + d], ak);
      }
      cur = sg; aq = 0.f; ak = 0.f;
    }
    aq += qv; ak += kv;
  }
  if (cur >= 0) {
    atomicAdd(&qlf[((size_t)bh * NL + cur) * ND + d], aq);
    atomicAdd(&klf[((size_t)bh * NL + cur) * ND + d], ak);
  }
}

// q_l carries the softmax scale 1/sqrt(64)=0.125 and the /64 landmark divisor;
// rows [65,96) zero-padded. grid (16 bh, 96 rows), block 64.
__global__ void cvt_landmarks_kernel(const float* __restrict__ qlf, const float* __restrict__ klf,
                                     __bf16* __restrict__ qlbf, __bf16* __restrict__ klbf)
{
  int bh = blockIdx.x, row = blockIdx.y, d = threadIdx.x;
  size_t o = ((size_t)bh * NLP + row) * ND + d;
  if (row < NL) {
    size_t s = ((size_t)bh * NL + row) * ND + d;
    qlbf[o] = (__bf16)(qlf[s] * (0.125f / 64.f));
    klbf[o] = (__bf16)(klf[s] * (1.f / 64.f));
  } else {
    qlbf[o] = (__bf16)0.f;
    klbf[o] = (__bf16)0.f;
  }
}

// ---------------------------------------------------------------------------
// sim1 = q @ k_l^T, fused softmax (scale 0.125) -> attn1 bf16 [bh][N][96]
// grid (N/128, 16), block 256. Wave = 16-row strip x 6 col tiles, K=64.
// Softmax done in registers with 16-lane shuffle reductions.
// ---------------------------------------------------------------------------
__global__ __launch_bounds__(256) void sim1_attn1_kernel(
    const __bf16* __restrict__ qkv, const __bf16* __restrict__ klbf, __bf16* __restrict__ attn1)
{
  __shared__ __bf16 Qt[128 * 72];
  __shared__ __bf16 Kt[NLP * 72];
  const int tid = threadIdx.x, lane = tid & 31, wid = tid >> 5;
  const int half = lane >> 4, lr = lane & 15;
  const int bh = blockIdx.y, b = bh >> 3, h = bh & 7;
  const int rbase = blockIdx.x * 128;

#pragma unroll
  for (int i = 0; i < 3; ++i) {           // k_l tile 96x64
    int c = tid + i * 256, r = c >> 3, cc = (c & 7) * 8;
    *(uint4*)&Kt[r * 72 + cc] = *(const uint4*)&klbf[(size_t)bh * NLP * ND + r * ND + cc];
  }
#pragma unroll
  for (int i = 0; i < 4; ++i) {           // q tile 128x64 (strided rows of qkv)
    int c = tid + i * 256, r = c >> 3, cc = (c & 7) * 8;
    int g = b * NN + rbase + r;
    *(uint4*)&Qt[r * 72 + cc] = *(const uint4*)&qkv[(size_t)g * (3 * NDIM) + h * ND + cc];
  }
  __syncthreads();

  v8f acc[6] = {};
#pragma unroll
  for (int k0 = 0; k0 < 64; k0 += 32) {
    int r = wid * 16 + lr;
    v16bf a = cat8(*(const v8bf*)&Qt[r * 72 + k0 + half * 8],
                   *(const v8bf*)&Qt[r * 72 + k0 + 16 + half * 8]);
#pragma unroll
    for (int t = 0; t < 6; ++t) {
      int n = t * 16 + lr;
      v16bf bm = cat8(*(const v8bf*)&Kt[n * 72 + k0 + half * 8],
                      *(const v8bf*)&Kt[n * 72 + k0 + 16 + half * 8]);
      acc[t] = wmma_bf16(a, bm, acc[t]);
    }
  }

  const float scale = 0.125f;
#pragma unroll
  for (int j = 0; j < 8; ++j) {
    float e[6];
    float m = -3.0e38f;
#pragma unroll
    for (int t = 0; t < 6; ++t) {
      int col = t * 16 + lr;
      float x = (col < NL) ? acc[t][j] * scale : -3.0e38f;
      e[t] = x;
      m = fmaxf(m, x);
    }
#pragma unroll
    for (int o = 8; o >= 1; o >>= 1) m = fmaxf(m, __shfl_xor(m, o, 16));
    float s = 0.f;
#pragma unroll
    for (int t = 0; t < 6; ++t) {
      int col = t * 16 + lr;
      float ev = (col < NL) ? __expf(e[t] - m) : 0.f;
      e[t] = ev; s += ev;
    }
#pragma unroll
    for (int o = 8; o >= 1; o >>= 1) s += __shfl_xor(s, o, 16);
    float inv = 1.f / s;
    int row = rbase + wid * 16 + half * 8 + j;
#pragma unroll
    for (int t = 0; t < 6; ++t) {
      int col = t * 16 + lr;
      attn1[(size_t)bh * NN * NLP + (size_t)row * NLP + col] = (__bf16)(e[t] * inv);
    }
  }
}

// ---------------------------------------------------------------------------
// sim3 = q_l @ k^T  (f32 out, [bh][80][N]); q_l already carries 0.125 scale.
// grid (N/128, 16), block 256. Wave = col tile x 5 row tiles.
// ---------------------------------------------------------------------------
__global__ __launch_bounds__(256) void sim3_kernel(
    const __bf16* __restrict__ qkv, const __bf16* __restrict__ qlbf, float* __restrict__ sim3f)
{
  __shared__ __bf16 Qt[NLP80 * 72];
  __shared__ __bf16 Kt[128 * 72];
  const int tid = threadIdx.x, lane = tid & 31, wid = tid >> 5;
  const int half = lane >> 4, lr = lane & 15;
  const int bh = blockIdx.y, b = bh >> 3, h = bh & 7;
  const int cbase = blockIdx.x * 128;

  for (int c = tid; c < NLP80 * 8; c += 256) {
    int r = c >> 3, cc = (c & 7) * 8;
    *(uint4*)&Qt[r * 72 + cc] = *(const uint4*)&qlbf[(size_t)bh * NLP * ND + r * ND + cc];
  }
#pragma unroll
  for (int i = 0; i < 4; ++i) {
    int c = tid + i * 256, r = c >> 3, cc = (c & 7) * 8;
    int g = b * NN + cbase + r;
    *(uint4*)&Kt[r * 72 + cc] = *(const uint4*)&qkv[(size_t)g * (3 * NDIM) + NDIM + h * ND + cc];
  }
  __syncthreads();

  v8f acc[5] = {};
#pragma unroll
  for (int k0 = 0; k0 < 64; k0 += 32) {
    int n = wid * 16 + lr;
    v16bf bm = cat8(*(const v8bf*)&Kt[n * 72 + k0 + half * 8],
                    *(const v8bf*)&Kt[n * 72 + k0 + 16 + half * 8]);
#pragma unroll
    for (int rt = 0; rt < 5; ++rt) {
      int r = rt * 16 + lr;
      v16bf a = cat8(*(const v8bf*)&Qt[r * 72 + k0 + half * 8],
                     *(const v8bf*)&Qt[r * 72 + k0 + 16 + half * 8]);
      acc[rt] = wmma_bf16(a, bm, acc[rt]);
    }
  }
#pragma unroll
  for (int rt = 0; rt < 5; ++rt)
#pragma unroll
    for (int j = 0; j < 8; ++j) {
      int m = rt * 16 + half * 8 + j;
      int col = cbase + wid * 16 + lr;
      sim3f[(size_t)bh * NLP80 * NN + (size_t)m * NN + col] = acc[rt][j];
    }
}

// Row softmax over N=16384 -> attn3 bf16; pad rows [65,80) written as zero.
__global__ __launch_bounds__(256) void softmax3_kernel(
    const float* __restrict__ sim3f, __bf16* __restrict__ attn3)
{
  __shared__ float red[256];
  int row = blockIdx.x, bh = blockIdx.y, tid = threadIdx.x;
  size_t base = (size_t)bh * NLP80 * NN + (size_t)row * NN;
  if (row >= NL) {
    for (int c = tid; c < NN; c += 256) attn3[base + c] = (__bf16)0.f;
    return;
  }
  float ev[64];
  float m = -3.0e38f;
#pragma unroll
  for (int i = 0; i < 64; ++i) {
    float x = sim3f[base + tid + i * 256];
    ev[i] = x; m = fmaxf(m, x);
  }
  red[tid] = m; __syncthreads();
  for (int o = 128; o >= 1; o >>= 1) { if (tid < o) red[tid] = fmaxf(red[tid], red[tid + o]); __syncthreads(); }
  m = red[0]; __syncthreads();
  float s = 0.f;
#pragma unroll
  for (int i = 0; i < 64; ++i) { ev[i] = __expf(ev[i] - m); s += ev[i]; }
  red[tid] = s; __syncthreads();
  for (int o = 128; o >= 1; o >>= 1) { if (tid < o) red[tid] += red[tid + o]; __syncthreads(); }
  float inv = 1.f / red[0];
#pragma unroll
  for (int i = 0; i < 64; ++i) attn3[base + tid + i * 256] = (__bf16)(ev[i] * inv);
}

// ---------------------------------------------------------------------------
// T3 = attn3 @ v : [80 x 16384] x [16384 x 64], split-K with f32 atomics.
// grid (32 k-splits, 16 bh), block 160 (5 waves); wave = 16-row strip.
// ---------------------------------------------------------------------------
__global__ __launch_bounds__(160) void t3_kernel(
    const __bf16* __restrict__ attn3, const __bf16* __restrict__ qkv, float* __restrict__ T3)
{
  __shared__ __bf16 At[NLP80 * 40];
  __shared__ __bf16 Vt[64 * 40];
  const int tid = threadIdx.x, lane = tid & 31, wid = tid >> 5;
  const int half = lane >> 4, lr = lane & 15;
  const int bh = blockIdx.y, b = bh >> 3, h = bh & 7;
  const int kbeg = blockIdx.x * 512;

  v8f acc[4] = {};
  for (int ks = 0; ks < 512; ks += 32) {
    int k0 = kbeg + ks;
    for (int c = tid; c < NLP80 * 4; c += 160) {
      int r = c >> 2, cc = (c & 3) * 8;
      *(uint4*)&At[r * 40 + cc] =
          *(const uint4*)&attn3[(size_t)bh * NLP80 * NN + (size_t)r * NN + k0 + cc];
    }
    for (int e = tid; e < 2048; e += 160) {     // transpose v rows into Vt[n][k]
      int r = e >> 6, col = e & 63;
      Vt[col * 40 + r] = qkv[(size_t)(b * NN + k0 + r) * (3 * NDIM) + 2 * NDIM + h * ND + col];
    }
    __syncthreads();
    int rr = wid * 16 + lr;
    v16bf a = cat8(*(const v8bf*)&At[rr * 40 + half * 8],
                   *(const v8bf*)&At[rr * 40 + 16 + half * 8]);
#pragma unroll
    for (int ct = 0; ct < 4; ++ct) {
      int n = ct * 16 + lr;
      v16bf bm = cat8(*(const v8bf*)&Vt[n * 40 + half * 8],
                      *(const v8bf*)&Vt[n * 40 + 16 + half * 8]);
      acc[ct] = wmma_bf16(a, bm, acc[ct]);
    }
    __syncthreads();
  }
#pragma unroll
  for (int ct = 0; ct < 4; ++ct)
#pragma unroll
    for (int j = 0; j < 8; ++j) {
      int m = wid * 16 + half * 8 + j;
      int c = ct * 16 + lr;
      atomicAdd(&T3[(size_t)bh * NLP80 * ND + m * ND + c], acc[ct][j]);
    }
}

// ---------------------------------------------------------------------------
// Per-(b,h): sim2 + softmax + 6 Moore-Penrose iterations + W = z @ T3.
// z_new = 3.25*Z - 0.25*(Z@P)@(15I - Q), Q = P@(7I-P), P = X@Z (algebraically
// identical to the reference). Dynamic LDS: 5 buffers of 65x66 f32.
// ---------------------------------------------------------------------------
static __device__ void mm65(const float* __restrict__ A, const float* __restrict__ B,
                            float* __restrict__ C, int tid) {
  for (int e = tid; e < NL * NL; e += 256) {
    int i = e / NL, j = e % NL;
    float s = 0.f;
    for (int k = 0; k < NL; ++k) s += A[i * 66 + k] * B[k * 66 + j];
    C[i * 66 + j] = s;
  }
  __syncthreads();
}

__global__ __launch_bounds__(256) void pinv_kernel(
    const float* __restrict__ qlf, const float* __restrict__ klf,
    const float* __restrict__ T3, __bf16* __restrict__ Wbf)
{
  extern __shared__ float sm[];
  float* X = sm;
  float* Z = X + NL * 66;
  float* P = Z + NL * 66;
  float* Q = P + NL * 66;
  float* U = Q + NL * 66;
  const int tid = threadIdx.x, bh = blockIdx.x;
  const float ss = 0.125f / (64.f * 64.f);
  const float* ql = qlf + (size_t)bh * NL * ND;
  const float* kl = klf + (size_t)bh * NL * ND;

  // sim2 logits
  for (int e = tid; e < NL * NL; e += 256) {
    int i = e / NL, j = e % NL;
    float s = 0.f;
    for (int d = 0; d < ND; ++d) s += ql[i * ND + d] * kl[j * ND + d];
    X[i * 66 + j] = s * ss;
  }
  __syncthreads();
  // row softmax -> attn2 (rows sum to 1 => max column-sum of abs(x) along -1 is 1)
  if (tid < NL) {
    float m = -3.0e38f;
    for (int j = 0; j < NL; ++j) m = fmaxf(m, X[tid * 66 + j]);
    float s = 0.f;
    for (int j = 0; j < NL; ++j) { float ev = __expf(X[tid * 66 + j] - m); X[tid * 66 + j] = ev; s += ev; }
    float inv = 1.f / s;
    for (int j = 0; j < NL; ++j) X[tid * 66 + j] *= inv;
  }
  __syncthreads();
  if (tid < NL) {                // column sums (the "row" abs-sum of the reference)
    float s = 0.f;
    for (int i = 0; i < NL; ++i) s += X[i * 66 + tid];
    U[tid] = s;
  }
  __syncthreads();
  float mx = 0.f;
  for (int j = 0; j < NL; ++j) mx = fmaxf(mx, U[j]);
  __syncthreads();
  for (int e = tid; e < NL * NL; e += 256) {   // z = x^T / (max(col)*max(row))
    int i = e / NL, j = e % NL;
    Z[i * 66 + j] = X[j * 66 + i] / mx;
  }
  __syncthreads();

  for (int it = 0; it < 6; ++it) {
    mm65(X, Z, P, tid);                        // P = X@Z
    for (int e = tid; e < NL * NL; e += 256) { // Q = P@(7I - P)
      int i = e / NL, j = e % NL;
      float s = 0.f;
      for (int k = 0; k < NL; ++k) s += P[i * 66 + k] * (((k == j) ? 7.f : 0.f) - P[k * 66 + j]);
      Q[i * 66 + j] = s;
    }
    __syncthreads();
    mm65(Z, P, U, tid);                        // U = Z@P
    for (int e = tid; e < NL * NL; e += 256) { // Znew = 3.25Z - 0.25*U@(15I - Q)  -> P
      int i = e / NL, j = e % NL;
      float s = 0.f;
      for (int k = 0; k < NL; ++k) s += U[i * 66 + k] * (((k == j) ? 15.f : 0.f) - Q[k * 66 + j]);
      P[i * 66 + j] = 3.25f * Z[i * 66 + j] - 0.25f * s;
    }
    __syncthreads();
    float* t = Z; Z = P; P = t;
  }

  // W = Z @ T3, padded to 96 rows (zeros)
  const float* t3 = T3 + (size_t)bh * NLP80 * ND;
  for (int e = tid; e < NLP * ND; e += 256) {
    int i = e >> 6, d = e & 63;
    float s = 0.f;
    if (i < NL)
      for (int k = 0; k < NL; ++k) s += Z[i * 66 + k] * t3[k * ND + d];
    Wbf[(size_t)bh * NLP * ND + e] = (__bf16)s;
  }
}

// ---------------------------------------------------------------------------
// Depthwise 33-tap conv residual over the (permuted) sequence, added to the
// attention output; emits bf16 A operand for the final projection GEMM.
// grid (32768, 2), block 256.
// ---------------------------------------------------------------------------
__global__ __launch_bounds__(256) void conv_res_kernel(
    const float* __restrict__ outh, const __bf16* __restrict__ qkv,
    const float* __restrict__ wres, __bf16* __restrict__ afin)
{
  int g = blockIdx.x;
  int c = blockIdx.y * 256 + threadIdx.x;  // c = h*64 + d
  int h = c >> 6;
  int b = g >> 14, n = g & (NN - 1);
  float acc = outh[(size_t)g * NDIM + c];
#pragma unroll
  for (int t = 0; t < 33; ++t) {
    int nn = n + t - 16;
    if (nn >= 0 && nn < NN)
      acc += wres[h * 33 + t] *
             (float)qkv[(size_t)((b << 14) + nn) * (3 * NDIM) + 2 * NDIM + c];
  }
  afin[(size_t)g * NDIM + c] = (__bf16)acc;
}

// ---------------------------------------------------------------------------
// Host-side orchestration
// ---------------------------------------------------------------------------
extern "C" void kernel_launch(void* const* d_in, const int* in_sizes, int n_in,
                              void* d_out, int out_size, void* d_ws, size_t ws_size,
                              hipStream_t stream) {
  const float* x         = (const float*)d_in[0];
  const int*   index     = (const int*)d_in[1];
  const int*   segids    = (const int*)d_in[3];
  const float* w_qkv     = (const float*)d_in[4];
  const float* w_out     = (const float*)d_in[5];
  const float* b_out     = (const float*)d_in[6];
  const float* w_res     = (const float*)d_in[7];
  float* out = (float*)d_out;
  char*  ws  = (char*)d_ws;

  size_t off = 0;
  auto take = [&](size_t bytes) { size_t r = off; off += (bytes + 255) & ~(size_t)255; return r; };
  __bf16* xp    = (__bf16*)(ws + take((size_t)NB * NN * NDIM * 2));
  __bf16* wqkvb = (__bf16*)(ws + take((size_t)NDIM * 3 * NDIM * 2));
  __bf16* woutb = (__bf16*)(ws + take((size_t)NDIM * NDIM * 2));
  __bf16* qkvb  = (__bf16*)(ws + take((size_t)NB * NN * 3 * NDIM * 2));
  float*  qlf   = (float*)(ws + take((size_t)16 * NL * ND * 4));     // these three are
  float*  klf   = (float*)(ws + take((size_t)16 * NL * ND * 4));     // contiguous (sizes
  float*  T3    = (float*)(ws + take((size_t)16 * NLP80 * ND * 4));  // multiples of 256B)
  __bf16* qlbf  = (__bf16*)(ws + take((size_t)16 * NLP * ND * 2));
  __bf16* klbf  = (__bf16*)(ws + take((size_t)16 * NLP * ND * 2));
  __bf16* attn1 = (__bf16*)(ws + take((size_t)16 * NN * NLP * 2));
  float*  sim3f = (float*)(ws + take((size_t)16 * NLP80 * NN * 4));
  __bf16* attn3 = (__bf16*)(ws + take((size_t)16 * NLP80 * NN * 2));
  __bf16* Wbf   = (__bf16*)(ws + take((size_t)16 * NLP * ND * 2));
  float*  outh  = (float*)(ws + take((size_t)NB * NN * NDIM * 4));
  __bf16* afin  = (__bf16*)(ws + take((size_t)NB * NN * NDIM * 2));

  // zero accumulators (qlf, klf, T3 are contiguous)
  {
    int n = 16 * NL * ND * 2 + 16 * NLP80 * ND;
    zero_f32_kernel<<<(n + 255) / 256, 256, 0, stream>>>(qlf, n);
  }
  cvt_f2bf_kernel<<<(NDIM * 3 * NDIM + 255) / 256, 256, 0, stream>>>(w_qkv, wqkvb, NDIM * 3 * NDIM);
  cvt_f2bf_kernel<<<(NDIM * NDIM + 255) / 256, 256, 0, stream>>>(w_out, woutb, NDIM * NDIM);
  gather_cvt_kernel<<<(NB * NN * NDIM) / 256, 256, 0, stream>>>(x, index, xp);

  // QKV projection: [32768 x 512] x [512 x 1536] -> bf16
  gemm_bf16_kernel<<<dim3(3 * NDIM / 64, NB * NN / 128, 1), 256, 0, stream>>>(
      xp, wqkvb, qkvb, NDIM, NDIM, 3 * NDIM, 3 * NDIM,
      0, 0, 0, 0, 0, 0, /*mode=*/0, nullptr, nullptr);

  landmarks_kernel<<<dim3(16, NN / 256), 256, 0, stream>>>(qkvb, segids, qlf, klf);
  cvt_landmarks_kernel<<<dim3(16, NLP), 64, 0, stream>>>(qlf, klf, qlbf, klbf);

  sim1_attn1_kernel<<<dim3(NN / 128, 16), 256, 0, stream>>>(qkvb, klbf, attn1);
  sim3_kernel<<<dim3(NN / 128, 16), 256, 0, stream>>>(qkvb, qlbf, sim3f);
  softmax3_kernel<<<dim3(NLP80, 16), 256, 0, stream>>>(sim3f, attn3);
  t3_kernel<<<dim3(32, 16), 160, 0, stream>>>(attn3, qkvb, T3);
  pinv_kernel<<<16, 256, 5 * NL * 66 * 4, stream>>>(qlf, klf, T3, Wbf);

  // out_heads = attn1 @ W, batched over 16 (b,h): z0 = b, z1 = h
  gemm_bf16_kernel<<<dim3(1, NN / 128, 16), 256, 0, stream>>>(
      attn1, Wbf, outh, NLP, NLP, ND, NDIM,
      8L * NN * NLP, (long)NN * NLP, 8L * NLP * ND, (long)NLP * ND,
      (long)NN * NDIM, 64L, /*mode=*/1, nullptr, nullptr);

  conv_res_kernel<<<dim3(NB * NN, 2), 256, 0, stream>>>(outh, qkvb, w_res, afin);

  // final projection + bias + unpermute scatter (final[index[j]] = permuted[j])
  gemm_bf16_kernel<<<dim3(NDIM / 64, NB * NN / 128, 1), 256, 0, stream>>>(
      afin, woutb, out, NDIM, NDIM, NDIM, NDIM,
      0, 0, 0, 0, 0, 0, /*mode=*/2, b_out, index);
}
